// OutlookAttention_18803366822524
// MI455X (gfx1250) — matrix operand order repaired and checked by reading, not compiled, not run on hardware
//
#include <hip/hip_runtime.h>
#include <hip/hip_bf16.h>

typedef __attribute__((ext_vector_type(16))) __bf16 v16bf;
typedef __attribute__((ext_vector_type(8)))  __bf16 v8bf;
typedef __attribute__((ext_vector_type(8)))  float  v8f;

#define HW    3136
#define CDIM  384
#define BATCH 8
#define MTOT  25088      // BATCH * HW ; 25088 = 49 * 512
#define NATT  976        // 972 attn logit channels padded to 61 tiles of 16
#define HGRID 56
#define KSLABS (CDIM / 32)   // 12 K-slabs of 32

// ---------------------------------------------------------------------------
// Build a 16-element bf16 fragment from two contiguous 16-byte chunks.
__device__ __forceinline__ v16bf ld16(const __bf16* p0, const __bf16* p1) {
  v8bf lo = *reinterpret_cast<const v8bf*>(p0);
  v8bf hi = *reinterpret_cast<const v8bf*>(p1);
  return __builtin_shufflevector(lo, hi, 0,1,2,3,4,5,6,7,8,9,10,11,12,13,14,15);
}

// Generic pointer -> 32-bit LDS byte offset (AS3->flat cast keeps offset in addr[31:0]).
__device__ __forceinline__ unsigned lds_off(const void* p) {
  return (unsigned)(unsigned long long)p;
}

// ---------------------------------------------------------------------------
// NT GEMM: C[M][N] = A[M][K=384] * Bw[N][K=384]^T (+bias), bf16 in, CT out.
// Block = 8 waves sharing one 16-col N-tile; weight tile (16x384 bf16, 12KB)
// staged into LDS fragment-ordered via global_load_async_to_lds_b128.
// Each wave computes a 64x16 tile with 4x v_wmma_f32_16x16x32_bf16 per K-slab.
template<typename CT>
__global__ void wmma_gemm_nt(const __bf16* __restrict__ A,
                             const __bf16* __restrict__ Bw,
                             const float*  __restrict__ bias,
                             CT* __restrict__ C,
                             int N) {
  __shared__ __align__(16) unsigned char sB[KSLABS * 1024];  // [slab][chunk0:512|chunk1:512]

  const int Ntiles = N >> 4;
  const int nt     = blockIdx.x % Ntiles;
  const int mblk   = blockIdx.x / Ntiles;
  const int waveId = threadIdx.x >> 5;
  const int lane   = threadIdx.x & 31;
  const int n0     = nt * 16;
  const long m0    = ((long)mblk * 8 + waveId) * 64;

  // ---- async-stage B tile: lane covers (row = lane&15, kb = (lane>>4)*16) ----
  {
    const int nrow = lane & 15, kb = (lane >> 4) * 16;
    const __bf16* src = Bw + (long)(n0 + nrow) * CDIM + kb;
    for (int kk = waveId; kk < KSLABS; kk += 8) {
      unsigned long long s0 = (unsigned long long)(const void*)(src + kk * 32);
      unsigned d0 = lds_off(sB + kk * 1024 + lane * 16);        // chunk0: K kb..kb+7
      unsigned d1 = d0 + 512;                                   // chunk1: K kb+8..kb+15
      asm volatile("global_load_async_to_lds_b128 %0, %1, off"
                   :: "v"(d0), "v"(s0) : "memory");
      asm volatile("global_load_async_to_lds_b128 %0, %1, off offset:16"
                   :: "v"(d1), "v"(s0) : "memory");
    }
    asm volatile("s_wait_asynccnt 0x0" ::: "memory");
  }
  __syncthreads();

  // ---- main K loop: A fragments straight from L2, B fragments from LDS ----
  const int rowA = lane & 15;
  const int kbA  = (lane >> 4) * 16;
  const __bf16* ap = A + (m0 + rowA) * CDIM + kbA;

  v8f acc0 = {}, acc1 = {}, acc2 = {}, acc3 = {};
  for (int k0 = 0; k0 < CDIM; k0 += 32) {
    __builtin_prefetch(ap + k0 + 64, 0, 1);
    const __bf16* bb = (const __bf16*)(sB + (k0 >> 5) * 1024) + lane * 8;
    v16bf bfrag = ld16(bb, bb + 256);                 // two conflict-free ds_load_b128
    v16bf a0 = ld16(ap + k0,           ap + k0 + 8);
    v16bf a1 = ld16(ap + 16*CDIM + k0, ap + 16*CDIM + k0 + 8);
    v16bf a2 = ld16(ap + 32*CDIM + k0, ap + 32*CDIM + k0 + 8);
    v16bf a3 = ld16(ap + 48*CDIM + k0, ap + 48*CDIM + k0 + 8);
    acc0 = __builtin_amdgcn_wmma_f32_16x16x32_bf16(false, a0, false, bfrag, (short)0, acc0, false, false);
    acc1 = __builtin_amdgcn_wmma_f32_16x16x32_bf16(false, a1, false, bfrag, (short)0, acc1, false, false);
    acc2 = __builtin_amdgcn_wmma_f32_16x16x32_bf16(false, a2, false, bfrag, (short)0, acc2, false, false);
    acc3 = __builtin_amdgcn_wmma_f32_16x16x32_bf16(false, a3, false, bfrag, (short)0, acc3, false, false);
  }

  const int cn  = n0 + (lane & 15);
  const int mlo = (lane >> 4) * 8;                    // D layout: vgpr r -> M=r (+8 hi lanes)
  float bv = bias ? bias[cn] : 0.0f;
  v8f accs[4] = {acc0, acc1, acc2, acc3};
  #pragma unroll
  for (int s = 0; s < 4; ++s) {
    #pragma unroll
    for (int r = 0; r < 8; ++r) {
      long row = m0 + s * 16 + r + mlo;
      C[row * (long)N + cn] = (CT)(accs[s][r] + bv);
    }
  }
}

// ---------------------------------------------------------------------------
// x [B][C][HW] f32  ->  xT [B*HW][C] bf16   (LDS 32x32 tiled transpose)
__global__ void x_to_bf16_t(const float* __restrict__ x, __bf16* __restrict__ xT) {
  __shared__ float t[32][33];
  const int l0 = blockIdx.x * 32, c0 = blockIdx.y * 32, b = blockIdx.z;
  const int tl = threadIdx.x & 31, tc = threadIdx.x >> 5;
  #pragma unroll
  for (int it = 0; it < 4; ++it) {
    int c = tc + it * 8;
    t[c][tl] = x[((long)b * CDIM + c0 + c) * HW + l0 + tl];
  }
  __syncthreads();
  #pragma unroll
  for (int it = 0; it < 4; ++it) {
    int l = tc + it * 8;
    xT[((long)b * HW + l0 + l) * CDIM + c0 + tl] = (__bf16)t[tl][l];
  }
}

// y [B*HW][C] f32 -> out [B][C][HW] f32 (final layout), LDS tiled
__global__ void y_transpose_out(const float* __restrict__ y, float* __restrict__ out) {
  __shared__ float t[32][33];
  const int l0 = blockIdx.x * 32, c0 = blockIdx.y * 32, b = blockIdx.z;
  const int tl = threadIdx.x & 31, tc = threadIdx.x >> 5;
  #pragma unroll
  for (int it = 0; it < 4; ++it) {
    int l = tc + it * 8;
    t[l][tl] = y[((long)b * HW + l0 + l) * CDIM + c0 + tl];
  }
  __syncthreads();
  #pragma unroll
  for (int it = 0; it < 4; ++it) {
    int c = tc + it * 8;
    out[((long)b * CDIM + c0 + c) * HW + l0 + tl] = t[tl][c];
  }
}

// f32 weights -> bf16, zero-padding elements beyond srcN
__global__ void cvt_pad(const float* __restrict__ src, __bf16* __restrict__ dst,
                        long srcN, long dstN) {
  long i = (long)blockIdx.x * blockDim.x + threadIdx.x;
  if (i >= dstN) return;
  dst[i] = (i < srcN) ? (__bf16)src[i] : (__bf16)0.0f;
}

// ---------------------------------------------------------------------------
// Softmax over d (9 taps) per (pixel, head n, window-pos i); channel layout
// o = (i*9+d)*12+n inside each pixel row of NATT.
__global__ void softmax9(const __bf16* __restrict__ a, __bf16* __restrict__ p) {
  long id = (long)blockIdx.x * blockDim.x + threadIdx.x;
  const long total = (long)MTOT * 108;        // 12 heads * 9 window positions
  if (id >= total) return;
  long pix = id / 108;
  int rem = (int)(id - pix * 108);
  int i = rem / 12, n = rem - i * 12;
  const __bf16* row = a + pix * NATT + n;
  float v[9]; float mx = -1e30f;
  #pragma unroll
  for (int d = 0; d < 9; ++d) { v[d] = (float)row[(i * 9 + d) * 12]; mx = fmaxf(mx, v[d]); }
  float s = 0.f;
  #pragma unroll
  for (int d = 0; d < 9; ++d) { v[d] = __expf(v[d] - mx); s += v[d]; }
  float inv = 1.0f / s;
  __bf16* prow = p + pix * NATT + n;
  #pragma unroll
  for (int d = 0; d < 9; ++d) prow[(i * 9 + d) * 12] = (__bf16)(v[d] * inv);
}

// ---------------------------------------------------------------------------
// Fused AV product + fold (gather form), vectorized over 8 head channels:
// thread = (b, pixel, head n, octet q). For each window position i=(di,dj) and
// tap d=(ki,kj):  acc[0..7] += P[b,(y+1-di,x+1-dj),i,d,n] * V[b,(y+ki-di,x+kj-dj),n*32+q*8..+7]
__global__ void fold_av(const __bf16* __restrict__ P, const __bf16* __restrict__ V,
                        __bf16* __restrict__ out) {
  long id = (long)blockIdx.x * blockDim.x + threadIdx.x;
  if (id >= (long)MTOT * 48) return;          // 12 heads * 4 octets per pixel
  int  q  = (int)(id & 3);
  long t  = id >> 2;
  int  n  = (int)(t % 12);
  long pl = t / 12;
  int  l  = (int)(pl % HW);
  int  b  = (int)(pl / HW);
  int  y  = l / HGRID, x = l - y * HGRID;
  int  cb = n * 32 + q * 8;

  float acc[8] = {0.f,0.f,0.f,0.f,0.f,0.f,0.f,0.f};
  #pragma unroll
  for (int di = 0; di < 3; ++di) {
    #pragma unroll
    for (int dj = 0; dj < 3; ++dj) {
      int yi = y + 1 - di, xi = x + 1 - dj;
      if (yi < 0 || yi >= HGRID || xi < 0 || xi >= HGRID) continue;
      const __bf16* prow = P + ((long)b * HW + yi * HGRID + xi) * NATT + n;
      int i = di * 3 + dj;
      #pragma unroll
      for (int ki = 0; ki < 3; ++ki) {
        #pragma unroll
        for (int kj = 0; kj < 3; ++kj) {
          int yv = y + ki - di, xv = x + kj - dj;
          if (yv < 0 || yv >= HGRID || xv < 0 || xv >= HGRID) continue;
          int d = ki * 3 + kj;
          float pw = (float)prow[(i * 9 + d) * 12];
          v8bf vv = *reinterpret_cast<const v8bf*>(
              V + ((long)b * HW + yv * HGRID + xv) * CDIM + cb);
          #pragma unroll
          for (int e = 0; e < 8; ++e) acc[e] += pw * (float)vv[e];
        }
      }
    }
  }
  v8bf o;
  #pragma unroll
  for (int e = 0; e < 8; ++e) o[e] = (__bf16)acc[e];
  *reinterpret_cast<v8bf*>(out + ((long)b * HW + l) * CDIM + cb) = o;
}

// ---------------------------------------------------------------------------
extern "C" void kernel_launch(void* const* d_in, const int* in_sizes, int n_in,
                              void* d_out, int out_size, void* d_ws, size_t ws_size,
                              hipStream_t stream) {
  const float* x      = (const float*)d_in[0];
  const float* v_w    = (const float*)d_in[1];
  const float* v_b    = (const float*)d_in[2];
  const float* attn_w = (const float*)d_in[3];
  const float* proj_w = (const float*)d_in[4];
  float* out = (float*)d_out;

  char* ws = (char*)d_ws;
  size_t off = 0;
  auto carve = [&](size_t bytes) -> void* {
    void* p = ws + off;
    off = (off + bytes + 255) & ~(size_t)255;
    return p;
  };
  __bf16* xT   = (__bf16*)carve((size_t)MTOT * CDIM * 2);   // x, pixel-major bf16
  __bf16* wv   = (__bf16*)carve((size_t)CDIM * CDIM * 2);   // v_w bf16
  __bf16* wa   = (__bf16*)carve((size_t)NATT * CDIM * 2);   // attn_w bf16, padded rows
  __bf16* wp   = (__bf16*)carve((size_t)CDIM * CDIM * 2);   // proj_w bf16
  __bf16* vT   = (__bf16*)carve((size_t)MTOT * CDIM * 2);   // v projection, pixel-major
  __bf16* aT   = (__bf16*)carve((size_t)MTOT * NATT * 2);   // attn logits
  __bf16* Pm   = (__bf16*)carve((size_t)MTOT * NATT * 2);   // softmax probs
  __bf16* preT = (__bf16*)carve((size_t)MTOT * CDIM * 2);   // folded AV, pixel-major
  float*  yF   = (float*) carve((size_t)MTOT * CDIM * 4);   // proj output, pixel-major

  const int T = 256;

  // 1) weight conversion (+ zero padding for attn rows 972..975)
  {
    long n = (long)CDIM * CDIM;
    cvt_pad<<<(int)((n + T - 1) / T), T, 0, stream>>>(v_w, wv, n, n);
    cvt_pad<<<(int)((n + T - 1) / T), T, 0, stream>>>(proj_w, wp, n, n);
    long ns = 972L * CDIM, nd = (long)NATT * CDIM;
    cvt_pad<<<(int)((nd + T - 1) / T), T, 0, stream>>>(attn_w, wa, ns, nd);
  }
  // 2) x -> pixel-major bf16
  x_to_bf16_t<<<dim3(HW / 32, CDIM / 32, BATCH), T, 0, stream>>>(x, xT);

  // 3) value projection GEMM (with bias): blocks = (MTOT/512) * Ntiles
  wmma_gemm_nt<__bf16><<<(MTOT / 512) * (CDIM / 16), T, 0, stream>>>(xT, wv, v_b, vT, CDIM);
  // 4) attention-logit GEMM
  wmma_gemm_nt<__bf16><<<(MTOT / 512) * (NATT / 16), T, 0, stream>>>(xT, wa, nullptr, aT, NATT);
  // 5) softmax over the 9 taps
  {
    long rows = (long)MTOT * 108;
    softmax9<<<(int)((rows + T - 1) / T), T, 0, stream>>>(aT, Pm);
  }
  // 6) fused AV + fold (gather), 8-wide over head channels
  {
    long n = (long)MTOT * 48;
    fold_av<<<(int)((n + T - 1) / T), T, 0, stream>>>(Pm, vT, preT);
  }
  // 7) projection GEMM -> f32 pixel-major
  wmma_gemm_nt<float><<<(MTOT / 512) * (CDIM / 16), T, 0, stream>>>(preT, wp, nullptr, yF, CDIM);
  // 8) transpose to [B, C, H, W]
  y_transpose_out<<<dim3(HW / 32, CDIM / 32, BATCH), T, 0, stream>>>(yF, out);
}
